// ContrastiveLossSimCLR_40870908788867
// MI455X (gfx1250) — matrix operand order repaired and checked by reading, not compiled, not run on hardware
//
#include <hip/hip_runtime.h>

typedef __attribute__((ext_vector_type(16))) _Float16 v16h;
typedef __attribute__((ext_vector_type(8)))  _Float16 v8h;
typedef __attribute__((ext_vector_type(4)))  _Float16 v4h;
typedef __attribute__((ext_vector_type(8)))  float    v8f;
typedef __attribute__((ext_vector_type(4)))  float    v4f;

#define NN 8192
#define DD 128
#define NCHUNK 16
#define JBLOCKS (NN / 16)               /* 512 j 16-row blocks               */
#define JB_PER_CHUNK (JBLOCKS / NCHUNK) /* 32 tiles per wave's j-chunk       */
#define IB_PAIRS (JBLOCKS / 2)          /* 256 i-block pairs (2 per wave)    */
#define LN2F 0.6931471805599453f
#define EPSN 1e-8f
/* sqrt((1/T) * log2(e)) with T=0.5 : folded into BOTH normalized matrices,
   so every dot product already carries the full exp2 scale factor.        */
#define SQRT_EXP_SCALE 1.69864464f

// ---------------------------------------------------------------------------
// Kernel 1: row-wise L2 normalize (torch eps clamp), scale by sqrt(exp-scale),
// convert to f16. One wave32 per row.
// ---------------------------------------------------------------------------
__global__ __launch_bounds__(128)
void normalize_f16_kernel(const float* __restrict__ pred,
                          const float* __restrict__ posi,
                          _Float16* __restrict__ zp,
                          _Float16* __restrict__ zq) {
  int wid  = (blockIdx.x * blockDim.x + threadIdx.x) >> 5;
  int lane = threadIdx.x & 31;
  const float* src;
  _Float16*    dst;
  int row;
  if (wid < NN) { src = pred; dst = zp; row = wid; }
  else          { src = posi; dst = zq; row = wid - NN; }

  v4f x = *(const v4f*)(src + (size_t)row * DD + lane * 4);
  float ss = x.x * x.x + x.y * x.y + x.z * x.z + x.w * x.w;
#pragma unroll
  for (int m = 16; m >= 1; m >>= 1) ss += __shfl_xor(ss, m, 32);
  float inv = SQRT_EXP_SCALE / fmaxf(sqrtf(ss), EPSN);

  v4h h;
  h.x = (_Float16)(x.x * inv);
  h.y = (_Float16)(x.y * inv);
  h.z = (_Float16)(x.z * inv);
  h.w = (_Float16)(x.w * inv);
  *(v4h*)(dst + (size_t)row * DD + lane * 4) = h;
}

// ---------------------------------------------------------------------------
// Helper: recompute ONE diagonal tile (i-block vs itself and vs zq) and store
// the per-row diagonal exp values. Runs twice per diag-owning wave (once per i-block), outside
// the hot loop, so its branches/butterflies cost nothing.
// ---------------------------------------------------------------------------
__device__ __forceinline__ void diag_tile(const _Float16* __restrict__ zp,
                                          const _Float16* __restrict__ zq,
                                          const v16h* afrag, int ibase, int lane,
                                          float* __restrict__ dpp,
                                          float* __restrict__ dpq) {
  int brow_l = lane & 15;
  int boff   = (lane < 16) ? 0 : 16;
  const _Float16* bp = zp + (size_t)(ibase + brow_l) * DD + boff;
  const _Float16* bq = zq + (size_t)(ibase + brow_l) * DD + boff;
  v8f cpp = {};
  v8f cpq = {};
#pragma unroll
  for (int kc = 0; kc < 4; ++kc) {
    v16h bfp = *(const v16h*)(bp + kc * 32);
    v16h bfq = *(const v16h*)(bq + kc * 32);
    cpp = __builtin_amdgcn_wmma_f32_16x16x32_f16(false, afrag[kc], false, bfp,
                                                 (short)0, cpp, false, false);
    cpq = __builtin_amdgcn_wmma_f32_16x16x32_f16(false, afrag[kc], false, bfq,
                                                 (short)0, cpq, false, false);
  }
#pragma unroll
  for (int r = 0; r < 8; ++r) {
    // C layout: VGPR r -> (M=r, N=lane) lanes 0-15; (M=r+8, N=lane-16) lanes
    // 16-31  =>  tile diagonal lives in lane r and lane r+24.
    bool sel = (lane == r) || (lane == r + 24);
    float d1 = sel ? __builtin_amdgcn_exp2f(cpp[r]) : 0.f;
    float d2 = sel ? __builtin_amdgcn_exp2f(cpq[r]) : 0.f;
#pragma unroll
    for (int m = 8; m >= 1; m >>= 1) {
      d1 += __shfl_xor(d1, m, 16);
      d2 += __shfl_xor(d2, m, 16);
    }
    if ((lane & 15) == 0) {
      int row = ibase + r + ((lane >> 4) << 3);
      dpp[row] = d1;
      dpq[row] = d2;
    }
  }
}

// ---------------------------------------------------------------------------
// Kernel 2: fused tile GEMM + exp2 + row-sum. One wave owns TWO i-blocks
// (A kept in registers -> 16 WMMAs per B-tile load) and one j-chunk of 32
// tiles. Branch-free hot loop; diagonal recomputed once afterwards.
// ---------------------------------------------------------------------------
__global__ __launch_bounds__(128)
void simclr_tiles_kernel(const _Float16* __restrict__ zp,
                         const _Float16* __restrict__ zq,
                         float* __restrict__ sumpp,
                         float* __restrict__ sumpq,
                         float* __restrict__ dpp,
                         float* __restrict__ dpq) {
  int wid   = (blockIdx.x * blockDim.x + threadIdx.x) >> 5;
  int lane  = threadIdx.x & 31;
  int pair  = wid >> 4;            // i-block pair 0..255
  int chunk = wid & (NCHUNK - 1);  // j-chunk 0..15
  int ib0   = pair * 2;
  int ib1   = ib0 + 1;
  int ibase0 = ib0 << 4;
  int ibase1 = ib1 << 4;

  // A fragments (16x32 f16): lane<16 holds K 0..15,
  // lane>=16 holds K 16..31 (contiguous, matches B); row = M = lane&15.
  int arow_l = lane & 15;
  int aoff   = (lane < 16) ? 0 : 16;
  v16h af0[4], af1[4];
#pragma unroll
  for (int kc = 0; kc < 4; ++kc) {
    const _Float16* a0 = zp + (size_t)(ibase0 + arow_l) * DD + kc * 32 + aoff;
    const _Float16* a1 = zp + (size_t)(ibase1 + arow_l) * DD + kc * 32 + aoff;
    v8h lo0 = *(const v8h*)a0, hi0 = *(const v8h*)(a0 + 8);
    v8h lo1 = *(const v8h*)a1, hi1 = *(const v8h*)(a1 + 8);
    v16h t0, t1;
#pragma unroll
    for (int i = 0; i < 8; ++i) {
      t0[i] = lo0[i]; t0[i + 8] = hi0[i];
      t1[i] = lo1[i]; t1[i + 8] = hi1[i];
    }
    af0[kc] = t0;
    af1[kc] = t1;
  }

  float app0[8], apq0[8], app1[8], apq1[8];
#pragma unroll
  for (int r = 0; r < 8; ++r) { app0[r] = 0.f; apq0[r] = 0.f; app1[r] = 0.f; apq1[r] = 0.f; }

  // B fragment (32x16 f16): lane = column n (= row of Z_j), contiguous 16
  // halfs; lane<16 -> K 0..15, lane>=16 -> K 16..31 of the chunk.
  int brow_l = lane & 15;
  int boff   = (lane < 16) ? 0 : 16;

  int jb0 = chunk * JB_PER_CHUNK;
  for (int t = 0; t < JB_PER_CHUNK; ++t) {
    int jb = jb0 + t;
    const _Float16* bp = zp + (size_t)(jb * 16 + brow_l) * DD + boff;
    const _Float16* bq = zq + (size_t)(jb * 16 + brow_l) * DD + boff;
    if (t + 1 < JB_PER_CHUNK) {               // global_prefetch_b8, keep near
      __builtin_prefetch(bp + 16 * DD, 0, 3);
      __builtin_prefetch(bq + 16 * DD, 0, 3);
    }

    v8f cpp0 = {}, cpq0 = {}, cpp1 = {}, cpq1 = {};
#pragma unroll
    for (int kc = 0; kc < 4; ++kc) {
      v16h bfp = *(const v16h*)(bp + kc * 32);
      v16h bfq = *(const v16h*)(bq + kc * 32);
      cpp0 = __builtin_amdgcn_wmma_f32_16x16x32_f16(false, af0[kc], false, bfp,
                                                    (short)0, cpp0, false, false);
      cpq0 = __builtin_amdgcn_wmma_f32_16x16x32_f16(false, af0[kc], false, bfq,
                                                    (short)0, cpq0, false, false);
      cpp1 = __builtin_amdgcn_wmma_f32_16x16x32_f16(false, af1[kc], false, bfp,
                                                    (short)0, cpp1, false, false);
      cpq1 = __builtin_amdgcn_wmma_f32_16x16x32_f16(false, af1[kc], false, bfq,
                                                    (short)0, cpq1, false, false);
    }

    // exp scale already folded into the f16 data: plain v_exp_f32 + add.
#pragma unroll
    for (int r = 0; r < 8; ++r) {
      app0[r] += __builtin_amdgcn_exp2f(cpp0[r]);
      apq0[r] += __builtin_amdgcn_exp2f(cpq0[r]);
      app1[r] += __builtin_amdgcn_exp2f(cpp1[r]);
      apq1[r] += __builtin_amdgcn_exp2f(cpq1[r]);
    }
  }

  // Butterfly reduce each 16-lane half; lane 0 -> row ibase+r, lane 16 ->
  // row ibase+r+8. Each (chunk,row) slot written exactly once (no atomics).
#pragma unroll
  for (int r = 0; r < 8; ++r) {
    float s0 = app0[r], q0 = apq0[r], s1 = app1[r], q1 = apq1[r];
#pragma unroll
    for (int m = 8; m >= 1; m >>= 1) {
      s0 += __shfl_xor(s0, m, 16);
      q0 += __shfl_xor(q0, m, 16);
      s1 += __shfl_xor(s1, m, 16);
      q1 += __shfl_xor(q1, m, 16);
    }
    if ((lane & 15) == 0) {
      int half = (lane >> 4) << 3;
      int row0 = ibase0 + r + half;
      int row1 = ibase1 + r + half;
      sumpp[(size_t)chunk * NN + row0] = s0;
      sumpq[(size_t)chunk * NN + row0] = q0;
      sumpp[(size_t)chunk * NN + row1] = s1;
      sumpq[(size_t)chunk * NN + row1] = q1;
    }
  }

  // Diagonal tiles (jb == ib0 and jb == ib1) live in chunk ib0>>5: recompute
  // them once, outside the hot loop.
  if ((ib0 >> 5) == chunk) {
    diag_tile(zp, zq, af0, ibase0, lane, dpp, dpq);
    diag_tile(zp, zq, af1, ibase1, lane, dpp, dpq);
  }
}

// ---------------------------------------------------------------------------
// Kernel 3: combine chunk partials, per-row loss, mean. Single block.
// ---------------------------------------------------------------------------
__global__ __launch_bounds__(1024)
void simclr_finalize_kernel(const float* __restrict__ sumpp,
                            const float* __restrict__ sumpq,
                            const float* __restrict__ dpp,
                            const float* __restrict__ dpq,
                            float* __restrict__ out) {
  __shared__ float red[1024];
  int tid = threadIdx.x;
  float acc = 0.f;
  for (int i = tid; i < NN; i += 1024) {
    float sp = 0.f, sq = 0.f;
#pragma unroll
    for (int c = 0; c < NCHUNK; ++c) {
      sp += sumpp[(size_t)c * NN + i];
      sq += sumpq[(size_t)c * NN + i];
    }
    float neg = sp - dpp[i] + sq;
    float pos = dpq[i];
    // -log(pos/neg) = ln2 * (log2(neg) - log2(pos))
    acc += __builtin_amdgcn_logf(neg) - __builtin_amdgcn_logf(pos);
  }
  red[tid] = acc;
  __syncthreads();
  for (int s = 512; s >= 1; s >>= 1) {
    if (tid < s) red[tid] += red[tid + s];
    __syncthreads();
  }
  if (tid == 0) out[0] = red[0] * (LN2F / (float)NN);
}

// ---------------------------------------------------------------------------
extern "C" void kernel_launch(void* const* d_in, const int* in_sizes, int n_in,
                              void* d_out, int out_size, void* d_ws, size_t ws_size,
                              hipStream_t stream) {
  const float* pred = (const float*)d_in[0];
  const float* posi = (const float*)d_in[1];

  char* ws = (char*)d_ws;
  _Float16* zp = (_Float16*)ws;                                   // 2 MB
  _Float16* zq = zp + (size_t)NN * DD;                            // 2 MB
  float* sumpp = (float*)(ws + 2u * (size_t)NN * DD * sizeof(_Float16));
  float* sumpq = sumpp + (size_t)NCHUNK * NN;                     // 512 KB each
  float* dpp   = sumpq + (size_t)NCHUNK * NN;                     // 32 KB
  float* dpq   = dpp + NN;                                        // 32 KB

  // 1) normalize + scale + f16 convert: 2*N rows, 4 waves / 128-thread block
  normalize_f16_kernel<<<(2 * NN) / 4, 128, 0, stream>>>(pred, posi, zp, zq);

  // 2) fused WMMA tiles: 256 i-pairs * 16 chunks = 4096 waves
  simclr_tiles_kernel<<<(IB_PAIRS * NCHUNK) / 4, 128, 0, stream>>>(
      zp, zq, sumpp, sumpq, dpp, dpq);

  // 3) finalize scalar loss
  simclr_finalize_kernel<<<1, 1024, 0, stream>>>(sumpp, sumpq, dpp, dpq,
                                                 (float*)d_out);
}
